// CliffordEPBottleneckV2_32109175505695
// MI455X (gfx1250) — matrix operand order repaired and checked by reading, not compiled, not run on hardware
//
#include <hip/hip_runtime.h>
#include <hip/hip_bf16.h>

// ---------------------------------------------------------------------------
// CliffordEPBottleneckV2 on gfx1250:
//   out = [ (x @ W_in + b_in) · (I ⊗ M^3) ] @ W_out + b_out
// with M = 0.99 I - 0.001 (W_e + W_e^T)  (3 EP steps folded analytically).
// Both big GEMMs run on V_WMMA_F32_16X16X32_BF16 (bf16 in, f32 accum).
// ---------------------------------------------------------------------------

typedef __attribute__((ext_vector_type(16))) __bf16 v16bf;
typedef __attribute__((ext_vector_type(8)))  float  v8f;

#define LDA 40   // padded LDS stride (bf16 elems) for A tiles
#define LDB 40   // padded LDS stride (bf16 elems) for B tiles (stored k-contig)

// Pack two f32 into two bf16 (RNE). Probe: native packed cvt builtin if
// declared; otherwise IR-level fptrunc via convertvector so the backend can
// select a hardware bf16 convert on gfx1250 if one exists.
__device__ __forceinline__ unsigned int pk2bf(float lo, float hi) {
#if __has_builtin(__builtin_amdgcn_cvt_pk_bf16_f32)
  typedef __attribute__((ext_vector_type(2))) __bf16 v2bf;
  union { v2bf v; unsigned int u; } cv;
  cv.v = __builtin_amdgcn_cvt_pk_bf16_f32(lo, hi);
  return cv.u;
#else
  typedef __attribute__((ext_vector_type(2))) float  v2f;
  typedef __attribute__((ext_vector_type(2))) __bf16 v2bf;
  union { v2bf v; unsigned int u; } cv;
  cv.v = __builtin_convertvector((v2f){lo, hi}, v2bf);
  return cv.u;
#endif
}

union Frag { uint4 q[2]; v16bf v; };

// ---------------------------------------------------------------------------
// M3 = M*M*M with M = 0.99 I - 0.001 (W_e + W_e^T); one block of 64 threads.
// ---------------------------------------------------------------------------
__global__ void m3_kernel(const float* __restrict__ W_e, float* __restrict__ M3) {
  __shared__ float M[64], M2[64];
  const int t = threadIdx.x;              // 0..63
  const int i = t >> 3, j = t & 7;
  float m = ((i == j) ? 0.99f : 0.0f) - 0.001f * (W_e[i * 8 + j] + W_e[j * 8 + i]);
  M[t] = m;
  __syncthreads();
  float s = 0.f;
#pragma unroll
  for (int k = 0; k < 8; ++k) s = fmaf(M[i * 8 + k], M[k * 8 + j], s);
  M2[t] = s;
  __syncthreads();
  float s3 = 0.f;
#pragma unroll
  for (int k = 0; k < 8; ++k) s3 = fmaf(M2[i * 8 + k], M[k * 8 + j], s3);
  M3[t] = s3;
}

// ---------------------------------------------------------------------------
// Blade mixing: H2[b, od*8 + j] = bf16( sum_i H[b, od*8 + i] * M3[i][j] )
// One thread per 8-element blade group.
// ---------------------------------------------------------------------------
__global__ __launch_bounds__(256) void ep_kernel(const float* __restrict__ H,
                                                 const float* __restrict__ M3,
                                                 unsigned short* __restrict__ H2) {
  __shared__ float sM[64];
  if (threadIdx.x < 64) sM[threadIdx.x] = M3[threadIdx.x];
  __syncthreads();
  const size_t g = (size_t)blockIdx.x * 256 + threadIdx.x;
  const float* hp = H + g * 8;
  float4 a = *(const float4*)hp;
  float4 b = *(const float4*)(hp + 4);
  float h[8] = {a.x, a.y, a.z, a.w, b.x, b.y, b.z, b.w};
  float y[8];
#pragma unroll
  for (int j = 0; j < 8; ++j) {
    float s = 0.f;
#pragma unroll
    for (int i = 0; i < 8; ++i) s = fmaf(h[i], sM[i * 8 + j], s);
    y[j] = s;
  }
  uint4 o;
  o.x = pk2bf(y[0], y[1]);
  o.y = pk2bf(y[2], y[3]);
  o.z = pk2bf(y[4], y[5]);
  o.w = pk2bf(y[6], y[7]);
  *(uint4*)(H2 + g * 8) = o;
}

// ---------------------------------------------------------------------------
// Tiled GEMM: C(MxN) = A(MxK) @ B(KxN) + bias, f32 out, f32 accumulate,
// bf16 operands via V_WMMA_F32_16X16X32_BF16.
//   block tile 128x128, BK=32, 256 threads = 8 waves in 4(m) x 2(n),
//   each wave computes 2x4 tiles of 16x16 -> 8 WMMAs per K-step.
// A global: f32 (A_BF16=false) or bf16 (true), row-major lda=K.
// B global: f32 row-major ldb=N; staged transposed (k-contiguous per column).
// Requires M%128==0, N%128==0, K%32==0 (true for all calls here).
// ---------------------------------------------------------------------------
template <bool A_BF16>
__global__ __launch_bounds__(256) void gemm_bf16_wmma(
    const void*  __restrict__ Ag, const float* __restrict__ Bg,
    const float* __restrict__ bias, float* __restrict__ Cg,
    int K, int N)
{
  __shared__ __align__(16) unsigned short sA[2][128 * LDA];
  __shared__ __align__(16) unsigned short sB[2][128 * LDB];

  const int t    = threadIdx.x;
  const int lane = t & 31;
  const int wid  = t >> 5;
  const int wm   = wid >> 1;          // 0..3  -> 32-row strip
  const int wn   = wid & 1;           // 0..1  -> 64-col strip
  const int r    = lane & 15;
  const int hlf  = lane >> 4;

  const int m0 = blockIdx.y * 128;
  const int n0 = blockIdx.x * 128;

  // staging work assignment (4 units per thread each for A and B)
  const int a_m   = t >> 3;           // 0..31 (+32*i)
  const int a_kc  = (t & 7) << 2;     // k chunk base 0..28
  const int b_n   = t & 127;          // column handled by this thread
  const int b_kq0 = t >> 7;           // 0/1 ; kq = b_kq0 + 2*i

  // ISA 7.12.2 fragment chunk offsets (bf16 elements)
  const int a_c0 = 8 * hlf,  a_c1 = 16 + 8 * hlf;   // A: K {0..7,16..23}/{8..15,24..31}
  const int b_c0 = 8 * hlf,  b_c1 = 16 + 8 * hlf;   // B: K {0..7,16..23}/{8..15,24..31}

  v8f acc[2][4] = {};

  float4 a_f[4];
  uint2  a_b[4];
  float  b_f[4][4];

  auto load_global = [&](int kt) {
#pragma unroll
    for (int i = 0; i < 4; ++i) {
      const size_t row = (size_t)(m0 + a_m + 32 * i);
      const size_t off = row * (size_t)K + (size_t)kt * 32 + a_kc;
      if (A_BF16) a_b[i] = *(const uint2*)((const unsigned short*)Ag + off);
      else        a_f[i] = *(const float4*)((const float*)Ag + off);
    }
#pragma unroll
    for (int i = 0; i < 4; ++i) {
      const int kq = b_kq0 + 2 * i;
      const size_t kbase = (size_t)kt * 32 + kq * 4;
#pragma unroll
      for (int rr = 0; rr < 4; ++rr)
        b_f[i][rr] = Bg[(kbase + rr) * (size_t)N + (n0 + b_n)];
    }
  };

  auto store_lds = [&](int buf) {
#pragma unroll
    for (int i = 0; i < 4; ++i) {
      uint2 v;
      if (A_BF16) v = a_b[i];
      else { v.x = pk2bf(a_f[i].x, a_f[i].y); v.y = pk2bf(a_f[i].z, a_f[i].w); }
      *(uint2*)&sA[buf][(a_m + 32 * i) * LDA + a_kc] = v;
    }
#pragma unroll
    for (int i = 0; i < 4; ++i) {
      const int kq = b_kq0 + 2 * i;
      uint2 v;
      v.x = pk2bf(b_f[i][0], b_f[i][1]);
      v.y = pk2bf(b_f[i][2], b_f[i][3]);
      *(uint2*)&sB[buf][b_n * LDB + kq * 4] = v;
    }
  };

  auto compute = [&](int buf) {
    Frag af[2], bf4[4];
#pragma unroll
    for (int tm = 0; tm < 2; ++tm) {
      const unsigned short* p = &sA[buf][(wm * 32 + tm * 16 + r) * LDA];
      af[tm].q[0] = *(const uint4*)(p + a_c0);
      af[tm].q[1] = *(const uint4*)(p + a_c1);
    }
#pragma unroll
    for (int tn = 0; tn < 4; ++tn) {
      const unsigned short* p = &sB[buf][(wn * 64 + tn * 16 + r) * LDB];
      bf4[tn].q[0] = *(const uint4*)(p + b_c0);
      bf4[tn].q[1] = *(const uint4*)(p + b_c1);
    }
#pragma unroll
    for (int tm = 0; tm < 2; ++tm)
#pragma unroll
      for (int tn = 0; tn < 4; ++tn)
        acc[tm][tn] = __builtin_amdgcn_wmma_f32_16x16x32_bf16(
            false, af[tm].v, false, bf4[tn].v,
            (short)0, acc[tm][tn], false, false);
  };

  const int KT = K >> 5;
  load_global(0);
  store_lds(0);
  for (int kt = 0; kt < KT; ++kt) {
    __syncthreads();                       // buffer kt&1 ready for all waves
    if (kt + 1 < KT) load_global(kt + 1);  // prefetch next tile during WMMA
    compute(kt & 1);
    if (kt + 1 < KT) store_lds((kt + 1) & 1);
  }

  // Epilogue: D layout lane=N (r), VGPR v = row v (+8 for upper lane half).
#pragma unroll
  for (int tm = 0; tm < 2; ++tm) {
#pragma unroll
    for (int tn = 0; tn < 4; ++tn) {
      const int n = n0 + wn * 64 + tn * 16 + r;
      const float bv = bias[n];
      const int mbase = m0 + wm * 32 + tm * 16 + hlf * 8;
#pragma unroll
      for (int v = 0; v < 8; ++v)
        Cg[(size_t)(mbase + v) * (size_t)N + n] = acc[tm][tn][v] + bv;
    }
  }
}

// ---------------------------------------------------------------------------
extern "C" void kernel_launch(void* const* d_in, const int* in_sizes, int n_in,
                              void* d_out, int out_size, void* d_ws, size_t ws_size,
                              hipStream_t stream) {
  (void)in_sizes; (void)n_in; (void)out_size; (void)ws_size;
  const float* x     = (const float*)d_in[0];   // (1024, 4096)
  const float* W_in  = (const float*)d_in[1];   // (4096, 16384)
  const float* b_in  = (const float*)d_in[2];   // (16384,)
  const float* W_e   = (const float*)d_in[3];   // (8, 8)
  const float* W_out = (const float*)d_in[4];   // (16384, 2048)
  const float* b_out = (const float*)d_in[5];   // (2048,)
  float* out = (float*)d_out;                   // (1024, 2048) f32

  const int B = 1024, IN = 4096, OUT = 2048, HID = 16384;

  float* M3          = (float*)d_ws;                                 // 64 f32
  float* H           = (float*)((char*)d_ws + 256);                  // 64 MB f32
  unsigned short* H2 = (unsigned short*)((char*)d_ws + 256 +
                          (size_t)B * HID * sizeof(float));          // 32 MB bf16

  m3_kernel<<<1, 64, 0, stream>>>(W_e, M3);

  gemm_bf16_wmma<false><<<dim3(HID / 128, B / 128), 256, 0, stream>>>(
      (const void*)x, W_in, b_in, H, IN, HID);

  ep_kernel<<<(B * OUT) / 256, 256, 0, stream>>>(H, M3, H2);

  gemm_bf16_wmma<true><<<dim3(OUT / 128, B / 128), 256, 0, stream>>>(
      (const void*)H2, W_out, b_out, out, HID, OUT);
}
